// GRUModel_12395275616886
// MI455X (gfx1250) — matrix operand order repaired and checked by reading, not compiled, not run on hardware
//
#include <hip/hip_runtime.h>
#include <math.h>

// ---------------------------------------------------------------------------
// GatedGraphConv (GGNN) on MI455X / gfx1250, wave32 + WMMA f16->f32.
//   per step: m = h@W ; agg = scatter_add(m[src]->dst) ; GRU(agg, h) -> h
// GEMMs run on v_wmma_f32_16x16x32_f16. Weights are pre-packed (f32->f16)
// into the wave32 B-fragment layout and staged block-wide into LDS via
// GLOBAL_LOAD_ASYNC_TO_LDS (ASYNCcnt) so all 8 waves of a block share one
// L2 fetch. Working set (~155 MB) is L2-resident (192 MB); the edge scatter
// uses coalesced float4 gathers + f32 atomics.
// ---------------------------------------------------------------------------

#define NNODES 100000
#define NEDGES 600000
#define CD     128
#define NL     3
#define NS     10

typedef __attribute__((ext_vector_type(16))) _Float16 v16h;
typedef __attribute__((ext_vector_type(8)))  float    v8f;
typedef int v4i __attribute__((__vector_size__(4 * sizeof(int))));

// packed-weight region sizes (elements of _Float16), per layer:
//   W:   10 steps * (8 tiles * 4 kchunks * 32 lanes * 16) = 163840
//   WiT: 24 tiles * 4 * 32 * 16                           =  49152
//   WhT:                                                  =  49152
#define PW_W_PER_STEP 16384
#define PW_W_REGION   163840
#define PW_G_REGION   49152
#define PW_PER_LAYER  262144   // 163840 + 2*49152

#if __has_builtin(__builtin_amdgcn_global_load_async_to_lds_b128) && \
    __has_builtin(__builtin_amdgcn_s_wait_asynccnt)
#define HAVE_ASYNC_LDS 1
#else
#define HAVE_ASYNC_LDS 0
#endif

// 16B global -> LDS stage; async (VGPR-bypassing, ASYNCcnt) when available.
__device__ __forceinline__ void stage16(const uint4* g, uint4* l) {
#if HAVE_ASYNC_LDS
  __builtin_amdgcn_global_load_async_to_lds_b128(
      (__attribute__((address_space(1))) v4i*)(void*)g,
      (__attribute__((address_space(3))) v4i*)(void*)l, 0, 0);
#else
  *l = *g;
#endif
}

__device__ __forceinline__ void stage_wait() {
#if HAVE_ASYNC_LDS
  __builtin_amdgcn_s_wait_asynccnt(0);
#endif
}

__device__ __forceinline__ v8f wmma16(v16h a, v16h b, v8f c) {
  // D = A(16x32 f16) * B(32x16 f16) + C(16x16 f32)
  return __builtin_amdgcn_wmma_f32_16x16x32_f16(
      /*neg_a=*/false, a, /*neg_b=*/false, b,
      /*c_mod=*/(short)0, c, /*reuse_a=*/false, /*reuse_b=*/false);
}

__device__ __forceinline__ float sigmoidf_(float x) {
  return 1.0f / (1.0f + __expf(-x));
}

// A-operand fragment: lane L holds row M = L&15, half = L>>4.
// v16h element e maps to K = kc*32 + half*16 + e.
__device__ __forceinline__ v16h load_a_frag(const float* __restrict__ h,
                                            int row, int kc, int half) {
  const float* p = h + (size_t)row * CD + kc * 32 + half * 16;
  float4 x0 = *(const float4*)(p);
  float4 x1 = *(const float4*)(p + 4);
  float4 y0 = *(const float4*)(p + 8);
  float4 y1 = *(const float4*)(p + 12);
  v16h a;
  a[0]  = (_Float16)x0.x; a[1]  = (_Float16)x0.y;
  a[2]  = (_Float16)x0.z; a[3]  = (_Float16)x0.w;
  a[4]  = (_Float16)x1.x; a[5]  = (_Float16)x1.y;
  a[6]  = (_Float16)x1.z; a[7]  = (_Float16)x1.w;
  a[8]  = (_Float16)y0.x; a[9]  = (_Float16)y0.y;
  a[10] = (_Float16)y0.z; a[11] = (_Float16)y0.w;
  a[12] = (_Float16)y1.x; a[13] = (_Float16)y1.y;
  a[14] = (_Float16)y1.z; a[15] = (_Float16)y1.w;
  return a;
}

// B fragment from LDS: fragment-major, 32 lanes * 16 f16 per fragment.
__device__ __forceinline__ v16h lds_b_frag(const _Float16* sB, int frag, int lane) {
  return *(const v16h*)(sB + ((size_t)frag * 32 + lane) * 16);
}

// ---------------------------------------------------------------------------
// Pack all weights (f32 -> f16, fragment-major B layout) into d_ws.
// B layout: lane = half*16+n ; element e -> B[kc*32 + half*16 + e][tile*16+n]
// ---------------------------------------------------------------------------
__global__ __launch_bounds__(256) void pack_weights_kernel(
    const float* __restrict__ W,    // [3][10][128][128]
    const float* __restrict__ wih,  // [3][384][128]
    const float* __restrict__ whh,  // [3][384][128]
    _Float16* __restrict__ pw) {
  int idx = blockIdx.x * blockDim.x + threadIdx.x;
  if (idx >= NL * PW_PER_LAYER) return;
  int l   = idx / PW_PER_LAYER;
  int rem = idx % PW_PER_LAYER;
  int e    = rem & 15;
  int lane = (rem >> 4) & 31;
  int half = lane >> 4;
  int n    = lane & 15;
  float val;
  if (rem < PW_W_REGION) {                       // W[s]: B[k][col] = W[k*128+col]
    int s  = rem / PW_W_PER_STEP;
    int fr = (rem % PW_W_PER_STEP) >> 9;         // tile*4 + kc
    int t  = fr >> 2, kc = fr & 3;
    int k  = kc * 32 + half * 16 + e;
    int col = t * 16 + n;
    val = W[(((size_t)l * NS + s) * CD + k) * CD + col];
  } else if (rem < PW_W_REGION + PW_G_REGION) {  // WiT: B[k][o] = wih[o*128+k]
    int r2 = rem - PW_W_REGION;
    int fr = r2 >> 9;
    int t  = fr >> 2, kc = fr & 3;
    int k  = kc * 32 + half * 16 + e;
    int o  = t * 16 + n;
    val = wih[((size_t)l * 3 * CD + o) * CD + k];
  } else {                                       // WhT: B[k][o] = whh[o*128+k]
    int r2 = rem - PW_W_REGION - PW_G_REGION;
    int fr = r2 >> 9;
    int t  = fr >> 2, kc = fr & 3;
    int k  = kc * 32 + half * 16 + e;
    int o  = t * 16 + n;
    val = whh[((size_t)l * 3 * CD + o) * CD + k];
  }
  pw[idx] = (_Float16)val;
}

// ---------------------------------------------------------------------------
// m = h @ W   (N x 128) @ (128 x 128).  8 waves/block, 16 rows/wave.
// W fragments (32 KB) staged once per block in LDS, shared by all waves.
// ---------------------------------------------------------------------------
__global__ __launch_bounds__(256) void gemm_hW_kernel(
    const float* __restrict__ h, const _Float16* __restrict__ pwW,
    float* __restrict__ m) {
  __shared__ _Float16 sW[32 * 512];              // 32 fragments = 32 KB

  // cooperative, fully-coalesced stage of all W fragments (async -> LDS)
  {
    const uint4* g = (const uint4*)pwW;          // 2048 uint4 total
    uint4* s = (uint4*)sW;
#pragma unroll
    for (int i = 0; i < 8; ++i)
      stage16(g + threadIdx.x + i * 256, s + threadIdx.x + i * 256);
    stage_wait();
  }
  __syncthreads();

  int lane = threadIdx.x & 31;
  int wave = threadIdx.x >> 5;
  int row0 = blockIdx.x * 128 + wave * 16;
  int half = lane >> 4;
  int mrow = row0 + (lane & 15);
  if (mrow >= NNODES) mrow = NNODES - 1;

  v16h a[4];
#pragma unroll
  for (int kc = 0; kc < 4; ++kc) a[kc] = load_a_frag(h, mrow, kc, half);

  int n = lane & 15;
  // block-uniform tail guard -> scalar branch, no exec-mask churn
  bool blockFull = ((blockIdx.x + 1) * 128 <= NNODES);
#pragma unroll
  for (int t = 0; t < 8; ++t) {
    v8f acc = {0.f, 0.f, 0.f, 0.f, 0.f, 0.f, 0.f, 0.f};
#pragma unroll
    for (int kc = 0; kc < 4; ++kc)
      acc = wmma16(a[kc], lds_b_frag(sW, t * 4 + kc, lane), acc);

    int col = t * 16 + n;
    if (blockFull) {
#pragma unroll
      for (int j = 0; j < 8; ++j)                // lanes 0-15: 64B contiguous
        m[(size_t)(row0 + j + 8 * half) * CD + col] = acc[j];
    } else {
#pragma unroll
      for (int j = 0; j < 8; ++j) {
        int r = row0 + j + 8 * half;
        if (r < NNODES) m[(size_t)r * CD + col] = acc[j];
      }
    }
  }
}

// ---------------------------------------------------------------------------
// agg zeroing + edge scatter-add (32 threads/edge, float4 gather, f32 atomics)
// ---------------------------------------------------------------------------
__global__ __launch_bounds__(256) void zero_kernel(float4* __restrict__ p, int n4) {
  int i = blockIdx.x * blockDim.x + threadIdx.x;
  if (i < n4) p[i] = make_float4(0.f, 0.f, 0.f, 0.f);
}

__global__ __launch_bounds__(256) void scatter_add_kernel(
    const float* __restrict__ m, const int* __restrict__ src,
    const int* __restrict__ dst, float* __restrict__ agg) {
  int tid = blockIdx.x * blockDim.x + threadIdx.x;
  int e  = tid >> 5;
  int c4 = (tid & 31) * 4;
  if (e >= NEDGES) return;
  int s = src[e], d = dst[e];
  float4 v = *(const float4*)(m + (size_t)s * CD + c4);
  float* o = agg + (size_t)d * CD + c4;
  atomicAdd(o + 0, v.x);
  atomicAdd(o + 1, v.y);
  atomicAdd(o + 2, v.z);
  atomicAdd(o + 3, v.w);
}

// ---------------------------------------------------------------------------
// Fused GRU gate: gi = agg@WiT+bi ; gh = h@WhT+bh ; gating -> h_new
// Per column tile, the 6 needed B chunks (24 KB) are staged in LDS and
// shared by all 8 waves; 24 WMMAs per tile per wave.
// ---------------------------------------------------------------------------
__global__ __launch_bounds__(256) void gru_gate_kernel(
    const float* __restrict__ agg, const float* __restrict__ hcur,
    const _Float16* __restrict__ pwi, const _Float16* __restrict__ pwh,
    const float* __restrict__ bi, const float* __restrict__ bh,
    float* __restrict__ hout, int apply_relu) {
  __shared__ _Float16 sB[24 * 512];              // 6 chunks * 4 kc = 24 KB

  int lane = threadIdx.x & 31;
  int wave = threadIdx.x >> 5;
  int row0 = blockIdx.x * 128 + wave * 16;
  int half = lane >> 4;
  int mrow = row0 + (lane & 15);
  if (mrow >= NNODES) mrow = NNODES - 1;

  v16h aA[4], aH[4];
#pragma unroll
  for (int kc = 0; kc < 4; ++kc) {
    aA[kc] = load_a_frag(agg,  mrow, kc, half);
    aH[kc] = load_a_frag(hcur, mrow, kc, half);
  }

  int n = lane & 15;
  bool blockFull = ((blockIdx.x + 1) * 128 <= NNODES);

  for (int t = 0; t < 8; ++t) {
    __syncthreads();                             // previous tile fully consumed
    {
      // 6 contiguous 4 KB chunks: wi{r,z,n}(tile g*8+t), wh{r,z,n}
      const uint4* srcs[6] = {
          (const uint4*)(pwi + (size_t)((0 * 8 + t) * 4) * 512),
          (const uint4*)(pwi + (size_t)((1 * 8 + t) * 4) * 512),
          (const uint4*)(pwi + (size_t)((2 * 8 + t) * 4) * 512),
          (const uint4*)(pwh + (size_t)((0 * 8 + t) * 4) * 512),
          (const uint4*)(pwh + (size_t)((1 * 8 + t) * 4) * 512),
          (const uint4*)(pwh + (size_t)((2 * 8 + t) * 4) * 512)};
      uint4* s = (uint4*)sB;
#pragma unroll
      for (int i = 0; i < 6; ++i)
        stage16(&srcs[i][threadIdx.x], &s[i * 256 + threadIdx.x]);
      if (t < 7) {                               // prefetch next tile's chunks
        __builtin_prefetch(pwi + (size_t)((t + 1) * 4) * 512 + threadIdx.x * 8, 0, 0);
        __builtin_prefetch(pwh + (size_t)((t + 1) * 4) * 512 + threadIdx.x * 8, 0, 0);
      }
      stage_wait();
    }
    __syncthreads();

    v8f ir = {0.f, 0.f, 0.f, 0.f, 0.f, 0.f, 0.f, 0.f};
    v8f iz = ir, in_ = ir, hr = ir, hz = ir, hn = ir;
#pragma unroll
    for (int kc = 0; kc < 4; ++kc) {
      ir  = wmma16(aA[kc], lds_b_frag(sB, 0 * 4 + kc, lane), ir);
      iz  = wmma16(aA[kc], lds_b_frag(sB, 1 * 4 + kc, lane), iz);
      in_ = wmma16(aA[kc], lds_b_frag(sB, 2 * 4 + kc, lane), in_);
      hr  = wmma16(aH[kc], lds_b_frag(sB, 3 * 4 + kc, lane), hr);
      hz  = wmma16(aH[kc], lds_b_frag(sB, 4 * 4 + kc, lane), hz);
      hn  = wmma16(aH[kc], lds_b_frag(sB, 5 * 4 + kc, lane), hn);
    }

    int col = t * 16 + n;
    float bir = bi[col], biz = bi[CD + col], bin = bi[2 * CD + col];
    float bhr = bh[col], bhz = bh[CD + col], bhn = bh[2 * CD + col];

    if (blockFull) {
#pragma unroll
      for (int j = 0; j < 8; ++j) {
        size_t r = (size_t)(row0 + j + 8 * half);
        float hv = hcur[r * CD + col];
        float rg = sigmoidf_(ir[j] + bir + hr[j] + bhr);
        float zg = sigmoidf_(iz[j] + biz + hz[j] + bhz);
        float ng = tanhf(in_[j] + bin + rg * (hn[j] + bhn));
        float hnew = (1.0f - zg) * ng + zg * hv;
        if (apply_relu) hnew = fmaxf(hnew, 0.0f);
        hout[r * CD + col] = hnew;
      }
    } else {
#pragma unroll
      for (int j = 0; j < 8; ++j) {
        int r = row0 + j + 8 * half;
        if (r < NNODES) {
          float hv = hcur[(size_t)r * CD + col];
          float rg = sigmoidf_(ir[j] + bir + hr[j] + bhr);
          float zg = sigmoidf_(iz[j] + biz + hz[j] + bhz);
          float ng = tanhf(in_[j] + bin + rg * (hn[j] + bhn));
          float hnew = (1.0f - zg) * ng + zg * hv;
          if (apply_relu) hnew = fmaxf(hnew, 0.0f);
          hout[(size_t)r * CD + col] = hnew;
        }
      }
    }
  }
}

// ---------------------------------------------------------------------------
// Host orchestration (graph-capture safe: kernels on `stream` only).
// ---------------------------------------------------------------------------
extern "C" void kernel_launch(void* const* d_in, const int* in_sizes, int n_in,
                              void* d_out, int out_size, void* d_ws, size_t ws_size,
                              hipStream_t stream) {
  (void)in_sizes; (void)n_in; (void)out_size; (void)ws_size;

  const float* x   = (const float*)d_in[0];
  const int*   ei  = (const int*)d_in[1];     // [2][NEDGES]
  const float* W   = (const float*)d_in[2];   // [3][10][128][128]
  const float* wih = (const float*)d_in[3];   // [3][384][128]
  const float* whh = (const float*)d_in[4];   // [3][384][128]
  const float* bih = (const float*)d_in[5];   // [3][384]
  const float* bhh = (const float*)d_in[6];   // [3][384]
  float* out = (float*)d_out;

  const size_t NODE_ELEMS = (size_t)NNODES * CD;          // 12.8M floats
  const size_t NODE_BYTES = ((NODE_ELEMS * 4 + 255) / 256) * 256;

  char* ws = (char*)d_ws;
  _Float16* pw  = (_Float16*)ws;                          // 1.5 MB packed weights
  float* bufA = (float*)(ws + (2u << 20));
  float* bufB = (float*)(ws + (2u << 20) + NODE_BYTES);
  float* aggb = (float*)(ws + (2u << 20) + 2 * NODE_BYTES);
  float* buf[2] = {bufA, bufB};

  const int* src = ei;
  const int* dst = ei + NEDGES;

  // pack all weights once per call
  {
    int total = NL * PW_PER_LAYER;                        // 786432
    pack_weights_kernel<<<(total + 255) / 256, 256, 0, stream>>>(W, wih, whh, pw);
  }

  const int ROW_BLOCKS = (NNODES + 127) / 128;            // 782
  const int ZERO_BLKS  = (int)(NODE_ELEMS / 4 / 256);     // 12500
  const int SCAT_BLKS  = (NEDGES * 32) / 256;             // 75000

  for (int g = 0; g < NL * NS; ++g) {
    int l = g / NS, s = g % NS;
    const float* hcur = (g == 0) ? x : buf[1 - (g & 1)];
    float* mbuf = buf[g & 1];
    float* hout = (g == NL * NS - 1) ? out : mbuf;        // m dead after scatter
    int relu = (s == NS - 1 && l < NL - 1) ? 1 : 0;

    const _Float16* pwW = pw + (size_t)l * PW_PER_LAYER + (size_t)s * PW_W_PER_STEP;
    const _Float16* pwi = pw + (size_t)l * PW_PER_LAYER + PW_W_REGION;
    const _Float16* pwh = pwi + PW_G_REGION;

    gemm_hW_kernel<<<ROW_BLOCKS, 256, 0, stream>>>(hcur, pwW, mbuf);
    zero_kernel<<<ZERO_BLKS, 256, 0, stream>>>((float4*)aggb, (int)(NODE_ELEMS / 4));
    scatter_add_kernel<<<SCAT_BLKS, 256, 0, stream>>>(mbuf, src, dst, aggb);
    gru_gate_kernel<<<ROW_BLOCKS, 256, 0, stream>>>(
        aggb, hcur, pwi, pwh, bih + (size_t)l * 3 * CD, bhh + (size_t)l * 3 * CD,
        hout, relu);
  }
}
